// GCN_47098611367945
// MI455X (gfx1250) — compile-verified, matrix operand order checked by reading, not run on hardware
//
#include <hip/hip_runtime.h>
#include <hip/hip_bf16.h>
#include <math.h>

#define N_NODES   100000
#define N_EDGES   3200000
#define F_IN      512
#define HID       16
#define N_CLASSES 40

typedef __attribute__((ext_vector_type(16))) __bf16 v16bf;
typedef __attribute__((ext_vector_type(8)))  float  v8f;

__device__ __forceinline__ __bf16 to_bf16(float f) { return (__bf16)f; }

// ---------------- degree / normalization ----------------
__global__ void k_init_deg(float* __restrict__ deg) {
  int i = blockIdx.x * blockDim.x + threadIdx.x;
  if (i < N_NODES) deg[i] = 1.0f;   // self-loop
}

__global__ void k_edge_deg(const int* __restrict__ dst, float* __restrict__ deg) {
  int e = blockIdx.x * blockDim.x + threadIdx.x;
  if (e < N_EDGES) atomicAdd(&deg[dst[e]], 1.0f);
}

__global__ void k_dinv(float* __restrict__ deg) {
  int i = blockIdx.x * blockDim.x + threadIdx.x;
  if (i < N_NODES) deg[i] = rsqrtf(deg[i]);
}

// ---------------- GEMM1: h1 = x @ W1  (bf16 WMMA, f32 accum) ----------------
// B (W1) is converted once per block into WMMA fragment layout in LDS:
// sB1[ks][lane][e], ks = k-step (K0 = ks*32), 16 bf16 per lane = one v16bf.
__global__ void k_gemm1(const float* __restrict__ x, const float* __restrict__ W1,
                        float* __restrict__ h1) {
  __shared__ __bf16 sB1[16 * 32 * 16];            // 16 KB
  for (int idx = threadIdx.x; idx < 16 * 32 * 16; idx += blockDim.x) {
    const int ks  = idx >> 9;                     // / 512
    const int rem = idx & 511;
    const int ln  = rem >> 4;
    const int e   = rem & 15;
    const int mm  = ln & 15;                      // B column
    const int hh  = ln >> 4;
    const int kk  = (e < 8) ? (hh * 8 + e) : (16 + hh * 8 + (e - 8));
    sB1[idx] = to_bf16(W1[(ks * 32 + kk) * HID + mm]);
  }
  __syncthreads();

  const int wave = threadIdx.x >> 5;
  const int lane = threadIdx.x & 31;
  const int tile = blockIdx.x * (blockDim.x >> 5) + wave;
  const int row0 = tile * 16;
  if (row0 < N_NODES) {                           // wave-uniform -> EXEC all-1 at WMMA
    const int m    = lane & 15;
    const int half = lane >> 4;
    const int koff = half * 8;
    const float* __restrict__ xrow = x + (size_t)(row0 + m) * F_IN;
    v8f acc = {};
    for (int ks = 0; ks < 16; ++ks) {
      const int k0 = ks * 32;
      const float4 x0 = *(const float4*)(xrow + k0 + koff);
      const float4 x1 = *(const float4*)(xrow + k0 + koff + 4);
      const float4 x2 = *(const float4*)(xrow + k0 + 16 + koff);
      const float4 x3 = *(const float4*)(xrow + k0 + 16 + koff + 4);
      v16bf a;
      a[0]  = to_bf16(x0.x); a[1]  = to_bf16(x0.y); a[2]  = to_bf16(x0.z); a[3]  = to_bf16(x0.w);
      a[4]  = to_bf16(x1.x); a[5]  = to_bf16(x1.y); a[6]  = to_bf16(x1.z); a[7]  = to_bf16(x1.w);
      a[8]  = to_bf16(x2.x); a[9]  = to_bf16(x2.y); a[10] = to_bf16(x2.z); a[11] = to_bf16(x2.w);
      a[12] = to_bf16(x3.x); a[13] = to_bf16(x3.y); a[14] = to_bf16(x3.z); a[15] = to_bf16(x3.w);
      const v16bf b = *(const v16bf*)&sB1[((ks << 5) + lane) << 4];
      acc = __builtin_amdgcn_wmma_f32_16x16x32_bf16(false, a, false, b,
                                                    (short)0, acc, false, false);
    }
    const int mbase = half * 8;                   // C/D: vgpr r -> M = mbase+r, N = m
#pragma unroll
    for (int r = 0; r < 8; ++r)
      h1[(size_t)(row0 + mbase + r) * HID + m] = acc[r];
  }
}

// ---------------- aggregation 1 (HID = 16) ----------------
__global__ void k_self1(const float* __restrict__ h1, const float* __restrict__ dinv,
                        float* __restrict__ agg1) {
  int idx = blockIdx.x * blockDim.x + threadIdx.x;
  if (idx >= N_NODES * HID) return;
  int n = idx >> 4;
  float w = dinv[n];
  agg1[idx] = w * w * h1[idx];                    // self-loop term, norm = dinv^2
}

__global__ void k_edge_agg1(const int* __restrict__ src, const int* __restrict__ dst,
                            const float* __restrict__ dinv, const float* __restrict__ h1,
                            float* __restrict__ agg1) {
  int idx = blockIdx.x * blockDim.x + threadIdx.x;
  if (idx >= N_EDGES * HID) return;
  int e = idx >> 4;
  int f = idx & 15;
  int s = src[e], d = dst[e];
  float w = dinv[s] * dinv[d];
  atomicAdd(&agg1[d * HID + f], w * h1[s * HID + f]);
}

__global__ void k_relu_bias(const float* __restrict__ agg1, const float* __restrict__ b1,
                            float* __restrict__ h) {
  int idx = blockIdx.x * blockDim.x + threadIdx.x;
  if (idx >= N_NODES * HID) return;
  float v = agg1[idx] + b1[idx & 15];
  h[idx] = v > 0.0f ? v : 0.0f;
}

// ---------------- GEMM2: h2 = h @ W2  (K=16 zero-padded to 32) ----------------
// B (W2) fragments staged in LDS with K-pad and N>=40 masking done at fill.
__global__ void k_gemm2(const float* __restrict__ h, const float* __restrict__ W2,
                        float* __restrict__ h2) {
  __shared__ __bf16 sB2[3 * 32 * 16];             // 3 KB
  for (int idx = threadIdx.x; idx < 3 * 32 * 16; idx += blockDim.x) {
    const int t   = idx >> 9;                     // N-tile
    const int rem = idx & 511;
    const int ln  = rem >> 4;
    const int e   = rem & 15;
    const int mm  = ln & 15;
    const int hh  = ln >> 4;
    const int n   = t * 16 + mm;
    float v = 0.0f;
    if (e < 8 && n < N_CLASSES) v = W2[(hh * 8 + e) * N_CLASSES + n];  // K = 0..15 real
    sB2[idx] = to_bf16(v);                        // K = 16..31 and n>=40 padded with 0
  }
  __syncthreads();

  const int wave = threadIdx.x >> 5;
  const int lane = threadIdx.x & 31;
  const int tile = blockIdx.x * (blockDim.x >> 5) + wave;
  const int row0 = tile * 16;
  if (row0 < N_NODES) {
    const int m    = lane & 15;
    const int half = lane >> 4;
    const int koff = half * 8;
    const float* __restrict__ hrow = h + (size_t)(row0 + m) * HID;
    const float4 h0 = *(const float4*)(hrow + koff);
    const float4 h1v = *(const float4*)(hrow + koff + 4);
    v16bf a;
    a[0] = to_bf16(h0.x);  a[1] = to_bf16(h0.y);  a[2] = to_bf16(h0.z);  a[3] = to_bf16(h0.w);
    a[4] = to_bf16(h1v.x); a[5] = to_bf16(h1v.y); a[6] = to_bf16(h1v.z); a[7] = to_bf16(h1v.w);
#pragma unroll
    for (int e = 8; e < 16; ++e) a[e] = to_bf16(0.0f);   // padded K = 16..31
    const int mbase = half * 8;
#pragma unroll
    for (int t = 0; t < 3; ++t) {                 // N tiles: 0-15, 16-31, 32-47
      const v16bf b = *(const v16bf*)&sB2[((t << 5) + lane) << 4];
      v8f acc = {};
      acc = __builtin_amdgcn_wmma_f32_16x16x32_bf16(false, a, false, b,
                                                    (short)0, acc, false, false);
      const int n = t * 16 + m;
      if (n < N_CLASSES) {
#pragma unroll
        for (int r = 0; r < 8; ++r)
          h2[(size_t)(row0 + mbase + r) * N_CLASSES + n] = acc[r];
      }
    }
  }
}

// ---------------- aggregation 2 (N_CLASSES = 40) ----------------
__global__ void k_self2(const float* __restrict__ h2, const float* __restrict__ dinv,
                        float* __restrict__ agg2) {
  int idx = blockIdx.x * blockDim.x + threadIdx.x;
  if (idx >= N_NODES * N_CLASSES) return;
  int n = idx / N_CLASSES;
  float w = dinv[n];
  agg2[idx] = w * w * h2[idx];
}

__global__ void k_edge_agg2(const int* __restrict__ src, const int* __restrict__ dst,
                            const float* __restrict__ dinv, const float* __restrict__ h2,
                            float* __restrict__ agg2) {
  int idx = blockIdx.x * blockDim.x + threadIdx.x;
  if (idx >= N_EDGES * N_CLASSES) return;
  int e = idx / N_CLASSES;
  int f = idx - e * N_CLASSES;
  int s = src[e], d = dst[e];
  float w = dinv[s] * dinv[d];
  atomicAdd(&agg2[d * N_CLASSES + f], w * h2[s * N_CLASSES + f]);
}

// ---------------- bias + log_softmax ----------------
__global__ void k_logsoftmax(const float* __restrict__ agg2, const float* __restrict__ b2,
                             float* __restrict__ out) {
  int i = blockIdx.x * blockDim.x + threadIdx.x;
  if (i >= N_NODES) return;
  const float* r = agg2 + (size_t)i * N_CLASSES;
  float v[N_CLASSES];
  float mx = -3.402823466e+38f;
#pragma unroll
  for (int f = 0; f < N_CLASSES; ++f) { v[f] = r[f] + b2[f]; mx = fmaxf(mx, v[f]); }
  float s = 0.0f;
#pragma unroll
  for (int f = 0; f < N_CLASSES; ++f) s += __expf(v[f] - mx);
  float ls = __logf(s);
#pragma unroll
  for (int f = 0; f < N_CLASSES; ++f) out[(size_t)i * N_CLASSES + f] = v[f] - mx - ls;
}

extern "C" void kernel_launch(void* const* d_in, const int* in_sizes, int n_in,
                              void* d_out, int out_size, void* d_ws, size_t ws_size,
                              hipStream_t stream) {
  (void)in_sizes; (void)n_in; (void)out_size; (void)ws_size;
  const float* x  = (const float*)d_in[0];
  const int*   ei = (const int*)d_in[1];
  const float* W1 = (const float*)d_in[2];
  const float* b1 = (const float*)d_in[3];
  const float* W2 = (const float*)d_in[4];
  const float* b2 = (const float*)d_in[5];
  const int* src = ei;               // edge_index[0]
  const int* dst = ei + N_EDGES;     // edge_index[1]
  float* out = (float*)d_out;

  // workspace carve-out (~45.2 MB)
  char* ws = (char*)d_ws;
  size_t off = 0;
  auto carve = [&](size_t bytes) -> float* {
    float* p = (float*)(ws + off);
    off += (bytes + 255) & ~(size_t)255;
    return p;
  };
  float* dinv = carve((size_t)N_NODES * 4);                 // deg -> dinv in place
  float* h1   = carve((size_t)N_NODES * HID * 4);           // x@W1, later relu(h)
  float* agg1 = carve((size_t)N_NODES * HID * 4);
  float* h2   = carve((size_t)N_NODES * N_CLASSES * 4);
  float* agg2 = carve((size_t)N_NODES * N_CLASSES * 4);

  const int TPB = 256;
  const int gN    = (N_NODES + TPB - 1) / TPB;
  const int gE    = (N_EDGES + TPB - 1) / TPB;
  const int gN16  = (N_NODES * HID + TPB - 1) / TPB;
  const int gE16  = (N_EDGES * HID + TPB - 1) / TPB;
  const int gN40  = (N_NODES * N_CLASSES + TPB - 1) / TPB;
  const int gE40  = (N_EDGES * N_CLASSES + TPB - 1) / TPB;
  const int tiles = (N_NODES + 15) / 16;                    // 6250
  const int gGEMM = (tiles + (TPB / 32) - 1) / (TPB / 32);  // 8 waves/block

  k_init_deg  <<<gN,    TPB, 0, stream>>>(dinv);
  k_edge_deg  <<<gE,    TPB, 0, stream>>>(dst, dinv);
  k_dinv      <<<gN,    TPB, 0, stream>>>(dinv);
  k_gemm1     <<<gGEMM, TPB, 0, stream>>>(x, W1, h1);
  k_self1     <<<gN16,  TPB, 0, stream>>>(h1, dinv, agg1);
  k_edge_agg1 <<<gE16,  TPB, 0, stream>>>(src, dst, dinv, h1, agg1);
  k_relu_bias <<<gN16,  TPB, 0, stream>>>(agg1, b1, h1);    // reuse h1 as relu(h)
  k_gemm2     <<<gGEMM, TPB, 0, stream>>>(h1, W2, h2);
  k_self2     <<<gN40,  TPB, 0, stream>>>(h2, dinv, agg2);
  k_edge_agg2 <<<gE40,  TPB, 0, stream>>>(src, dst, dinv, h2, agg2);
  k_logsoftmax<<<gN,    TPB, 0, stream>>>(agg2, b2, out);
}
